// XLALowRank_37185826848785
// MI455X (gfx1250) — matrix operand order repaired and checked by reading, not compiled, not run on hardware
//
#include <hip/hip_runtime.h>
#include <hip/hip_bf16.h>

// ---- problem constants (from reference) ----
enum : int { Bn = 16, Mn = 2048, En = 1024, Hn = 8, Dn = 128, MIDn = 64 };
#define EPS_GN 1e-3f

// gfx1250 async global->LDS copies for the B-tile staging (pure bf16 memcpy).
#define USE_ASYNC_LDS 1

typedef __bf16 bf16_t;
typedef __attribute__((ext_vector_type(16))) __bf16 v16bf;
typedef __attribute__((ext_vector_type(8)))  float  v8f;

// D = A(16x32 bf16) * B(32x16 bf16) + C (f32), wave32 WMMA
__device__ __forceinline__ v8f wmma_bf16(v16bf a, v16bf b, v8f c) {
    return __builtin_amdgcn_wmma_f32_16x16x32_bf16(
        false, a, false, b, (short)0, c, false, false);
}

// A-operand K permutation within a 32-wide chunk: lane lh=0 owns K{0-7,16-23},
// lh=1 owns K{8-15,24-31}; store so each lane's 16 elems are contiguous 32B.
__device__ __forceinline__ int permA(int k) {
    return (k & 7) | ((k & 8) << 1) | ((k & 16) >> 1);
}

// Fragment loads: one 32-byte LDS read per lane (2x ds_load_b128).
__device__ __forceinline__ v16bf load_afrag(const bf16_t* rowbase, int lh) {
    return *(const v16bf*)(rowbase + lh * 16);   // A: permuted-K layout
}
__device__ __forceinline__ v16bf load_bfrag(const bf16_t* colbase_p, int lh) {
    typedef __attribute__((ext_vector_type(8))) __bf16 v8bf;
    v8bf lo = *(const v8bf*)(colbase_p + lh * 8);      // B: [col][k], K{lh*8..+7}
    v8bf hi = *(const v8bf*)(colbase_p + 16 + lh * 8); //    and K{16+lh*8..+7}
    return __builtin_shufflevector(lo, hi, 0, 1, 2, 3, 4, 5, 6, 7, 8, 9, 10, 11, 12, 13, 14, 15);
}

__device__ __forceinline__ float elu1(float x) {
    return x > 0.f ? x : (__expf(x) - 1.f);
}

// pack 8 f32 (two float4) -> 8 bf16, one 16B vector store
__device__ __forceinline__ void pack8_store(float4 a, float4 b, bf16_t* dst) {
    union { bf16_t h[8]; uint4 v; } u;
    u.h[0] = (bf16_t)a.x; u.h[1] = (bf16_t)a.y; u.h[2] = (bf16_t)a.z; u.h[3] = (bf16_t)a.w;
    u.h[4] = (bf16_t)b.x; u.h[5] = (bf16_t)b.y; u.h[6] = (bf16_t)b.z; u.h[7] = (bf16_t)b.w;
    *(uint4*)dst = u.v;
}

// ---- async global -> LDS 16B copy (ASYNCcnt-tracked on gfx1250) ----
__device__ __forceinline__ void gl_async_b128(void* ldsdst, const void* gsrc) {
#if USE_ASYNC_LDS
    // LDS byte address = low 32 bits of the flat address (ISA flat->LDS mapping)
    asm volatile("global_load_async_to_lds_b128 %0, %1, off"
                 :: "v"((unsigned)(unsigned long long)ldsdst), "v"(gsrc)
                 : "memory");
#else
    *(uint4*)ldsdst = *(const uint4*)gsrc;
#endif
}
__device__ __forceinline__ void wait_async0() {
#if USE_ASYNC_LDS
    asm volatile("s_wait_asynccnt 0" ::: "memory");
#endif
}

// ---------------------------------------------------------------------------
// weight conversion: W (f32, K x N row-major) -> WT (bf16, [n][k] transposed)
// ---------------------------------------------------------------------------
__global__ __launch_bounds__(256) void cvt_T_bf16_kernel(const float* __restrict__ s,
                                                         bf16_t* __restrict__ d,
                                                         int K, int N) {
    int i = blockIdx.x * 256 + threadIdx.x;
    if (i < K * N) {
        int n = i % N, k = i / N;               // consecutive threads: coalesced read
        d[(size_t)n * K + k] = (bf16_t)s[(size_t)k * N + n];
    }
}

// ---------------------------------------------------------------------------
// per-b mask counts + zero pool accumulator
// ---------------------------------------------------------------------------
__global__ __launch_bounds__(256) void init_kernel(const float* __restrict__ mask,
                                                   float* __restrict__ pool_sum,
                                                   float* __restrict__ cnt) {
    int b = blockIdx.x, tid = threadIdx.x;
    __shared__ float red[256];
    float s = 0.f;
    for (int m = tid; m < Mn; m += 256) s += mask[b * Mn + m];
    red[tid] = s;
    __syncthreads();
    for (int o = 128; o > 0; o >>= 1) {
        if (tid < o) red[tid] += red[tid + o];
        __syncthreads();
    }
    if (tid == 0) cnt[b] = red[0];
    for (int i = tid; i < Hn * MIDn; i += 256) pool_sum[b * Hn * MIDn + i] = 0.f;
}

// ---------------------------------------------------------------------------
// staging helpers
//   A : ROWS x 32 f32->bf16, permuted-K layout, stride 40
//   B : 128 cols x 32 k, [col][k] from transposed bf16 weights, stride 40
// thread t: row/col = t>>1, half = t&1 covers elems half*16 .. half*16+15
// ---------------------------------------------------------------------------
__device__ __forceinline__ void stage_A32_full(bf16_t* As, const float* __restrict__ X,
                                               int r0, int kk, int tid) {
    int row = tid >> 1, half = tid & 1;   // 128 rows x 2 halves == 256 threads
    const float4* Xr = (const float4*)(X + (size_t)(r0 + row) * En + kk + half * 16);
    float4 a0 = Xr[0], a1 = Xr[1], a2 = Xr[2], a3 = Xr[3];
    bf16_t* dst = As + row * 40;
    pack8_store(a0, a1, dst + (half ? 8 : 0));    // K half*16+0..7
    pack8_store(a2, a3, dst + (half ? 24 : 16));  // K half*16+8..15
}
__device__ __forceinline__ void stage_B32_async(bf16_t* Bs, const bf16_t* __restrict__ WT,
                                                int colbase, int kk, int tid) {
    int col = tid >> 1, half = tid & 1;
    const bf16_t* src = WT + (size_t)(colbase + col) * En + kk + half * 16;
    bf16_t* dst = Bs + col * 40 + half * 16;
    gl_async_b128(dst, src);          // 16 bf16 = 32B per thread (2 x b128)
    gl_async_b128(dst + 8, src + 8);
}
__device__ __forceinline__ void stage_B32_sync(bf16_t* Bs, const bf16_t* __restrict__ WT,
                                               int colbase, int kk, int tid) {
    int col = tid >> 1, half = tid & 1;
    const bf16_t* src = WT + (size_t)(colbase + col) * En + kk + half * 16;
    bf16_t* dst = Bs + col * 40 + half * 16;
    *(uint4*)dst = *(const uint4*)src;
    *(uint4*)(dst + 8) = *(const uint4*)(src + 8);
}

// ---------------------------------------------------------------------------
// q / v1 projection: (16 x 1024) @ (1024 x 1024) + bias -> ELU -> GroupNorm
// grid.x = 16: [0..7]=q head h, [8..15]=v1 head h. 256 thr = 8 waves,
// wave w owns C tile (16 rows x cols w*16..w*16+15).
// ---------------------------------------------------------------------------
__global__ __launch_bounds__(256) void proj_small_kernel(
    const float* __restrict__ Xq, const float* __restrict__ Xv,
    const bf16_t* __restrict__ WqT, const bf16_t* __restrict__ WvT,
    const float* __restrict__ bq, const float* __restrict__ bv,
    const float* __restrict__ gq, const float* __restrict__ gv,
    const float* __restrict__ Bq, const float* __restrict__ Bv,
    float* __restrict__ qout, float* __restrict__ vout) {

    const int sel = blockIdx.x >> 3;
    const int h = blockIdx.x & 7;
    const float*  X    = sel ? Xv : Xq;
    const bf16_t* WT   = sel ? WvT : WqT;
    const float*  bias = sel ? bv : bq;
    const float*  gam  = sel ? gv : gq;
    const float*  bet  = sel ? Bv : Bq;
    float*        out  = sel ? vout : qout;

    const int tid = threadIdx.x;
    const int lane = tid & 31, w = tid >> 5;
    const int lm = lane & 15, lh = lane >> 4;
    const int colbase = h * Dn;

    __shared__ bf16_t As[16 * 40];
    __shared__ bf16_t Bs[128 * 40];
    __shared__ float  Cf[16 * 132];
    __shared__ float  mu_s[16], rs_s[16];

    v8f acc;
#pragma unroll
    for (int i = 0; i < 8; ++i) acc[i] = 0.f;

    for (int kk = 0; kk < En; kk += 32) {
        if (tid < 32) {  // 16 rows x 2 halves
            int row = tid >> 1, half = tid & 1;
            const float4* Xr = (const float4*)(X + (size_t)row * En + kk + half * 16);
            float4 a0 = Xr[0], a1 = Xr[1], a2 = Xr[2], a3 = Xr[3];
            bf16_t* dst = As + row * 40;
            pack8_store(a0, a1, dst + (half ? 8 : 0));
            pack8_store(a2, a3, dst + (half ? 24 : 16));
        }
        stage_B32_sync(Bs, WT, colbase, kk, tid);
        __syncthreads();
        v16bf a = load_afrag(As + lm * 40, lh);
        v16bf bb_ = load_bfrag(Bs + (w * 16 + lm) * 40, lh);
        acc = wmma_bf16(a, bb_, acc);
        __syncthreads();
    }

    float bcol = bias[colbase + w * 16 + lm];
#pragma unroll
    for (int r = 0; r < 8; ++r) {
        float x = elu1(acc[r] + bcol);
        acc[r] = x;
        Cf[(8 * lh + r) * 132 + w * 16 + lm] = x;
    }
    __syncthreads();
    if (tid < 16) {
        float s = 0.f, sq = 0.f;
        for (int c = 0; c < Dn; ++c) {
            float x = Cf[tid * 132 + c];
            s += x; sq += x * x;
        }
        float mu = s * (1.f / Dn);
        mu_s[tid] = mu;
        rs_s[tid] = rsqrtf(sq * (1.f / Dn) - mu * mu + EPS_GN);
    }
    __syncthreads();
    int col = colbase + w * 16 + lm;
    float g = gam[col], be = bet[col];
#pragma unroll
    for (int r = 0; r < 8; ++r) {
        int row = 8 * lh + r;
        out[row * En + col] = (acc[r] - mu_s[row]) * rs_s[row] * g + be;
    }
}

// ---------------------------------------------------------------------------
// Big projection: 128 rows x 128 cols tile per block, K=1024, bf16 WMMA,
// double-buffered LDS staging (A via VGPR cvt path, B via async-to-LDS).
// FUSED=true  (key): GEMM -> ELU -> GN -> *q -> relu(att@Wb+bb) -> logits+pool
// FUSED=false (value2): GEMM -> ELU -> GN -> store v2_gn
// ---------------------------------------------------------------------------
template <bool FUSED>
__global__ __launch_bounds__(256) void proj_big_kernel(
    const float* __restrict__ X,      // (Bn*Mn, En)
    const bf16_t* __restrict__ WT,    // (En, En) transposed [n][k] bf16
    const float* __restrict__ bias,   // En
    const float* __restrict__ gamma,  // En
    const float* __restrict__ beta,   // En
    const float* __restrict__ qg,     // (Bn, En)        [FUSED]
    const float* __restrict__ mask,   // (Bn, Mn)        [FUSED]
    const bf16_t* __restrict__ WbT,   // (MIDn, Dn) transposed [n][k]  [FUSED]
    const float* __restrict__ bbv,    // MIDn            [FUSED]
    const float* __restrict__ Wl1,    // MIDn            [FUSED]
    const float* __restrict__ bl1,    // 1               [FUSED]
    float* __restrict__ alpha_out,    // (Bn*Hn, Mn)     [FUSED]
    float* __restrict__ pool_sum,     // (Bn*Hn, MIDn)   [FUSED]
    float* __restrict__ vout) {       // (Bn*Mn, En)     [!FUSED]

    extern __shared__ char smem[];
    // region1: att bf16 128x136 (34816 B), A-perm layout; Lb f32 aliases after
    bf16_t* att = (bf16_t*)smem;
    float*  Lb  = (float*)smem;
    // region2: double-buffered As/Bs (4 x 10240 B); Wbs aliases after GEMM1
    bf16_t* As0 = (bf16_t*)(smem + 34816);
    bf16_t* As1 = (bf16_t*)(smem + 34816 + 10240);
    bf16_t* Bs0 = (bf16_t*)(smem + 34816 + 20480);
    bf16_t* Bs1 = (bf16_t*)(smem + 34816 + 30720);
    bf16_t* Wbs = (bf16_t*)(smem + 34816);  // 64x136 bf16 = 17408 <= 40960

    const int tid = threadIdx.x;
    const int lane = tid & 31, w = tid >> 5;
    const int lm = lane & 15, lh = lane >> 4;
    const int mrow = w * 16;
    const int r0 = blockIdx.x * 128;
    const int h = blockIdx.y;
    const int colbase = h * Dn;
    const int b = r0 >> 11;       // Mn = 2048
    const int m0 = r0 & (Mn - 1);

    v8f acc[8];
#pragma unroll
    for (int t = 0; t < 8; ++t)
#pragma unroll
        for (int i = 0; i < 8; ++i) acc[t][i] = 0.f;

    // ---- GEMM1: C = X[r0:r0+128, :] @ W[:, colbase:colbase+128] ----
    // software pipeline: stage chunk c+1 while computing chunk c
    stage_A32_full(As0, X, r0, 0, tid);
    stage_B32_async(Bs0, WT, colbase, 0, tid);
    wait_async0();
    __syncthreads();

    const int NK = En / 32;  // 32 chunks
    for (int c = 0; c < NK; ++c) {
        bf16_t* Ac = (c & 1) ? As1 : As0;
        bf16_t* Bc = (c & 1) ? Bs1 : Bs0;
        if (c + 1 < NK) {
            bf16_t* An = (c & 1) ? As0 : As1;
            bf16_t* Bn = (c & 1) ? Bs0 : Bs1;
            stage_A32_full(An, X, r0, (c + 1) * 32, tid);
            stage_B32_async(Bn, WT, colbase, (c + 1) * 32, tid);
        }
        v16bf a = load_afrag(Ac + (mrow + lm) * 40, lh);
#pragma unroll
        for (int t = 0; t < 8; ++t) {
            v16bf bfrag = load_bfrag(Bc + (t * 16 + lm) * 40, lh);
            acc[t] = wmma_bf16(a, bfrag, acc[t]);
        }
        wait_async0();      // next-chunk async B copies landed
        __syncthreads();    // + everyone done reading current buffers
    }

    // ---- bias + ELU ----
#pragma unroll
    for (int t = 0; t < 8; ++t) {
        float bcol = bias[colbase + t * 16 + lm];
#pragma unroll
        for (int r = 0; r < 8; ++r) acc[t][r] = elu1(acc[t][r] + bcol);
    }

    // ---- GroupNorm row stats (wave owns full 16x128 strip) ----
    float mean[8], rstd[8];
#pragma unroll
    for (int r = 0; r < 8; ++r) {
        float s = 0.f, sq = 0.f;
#pragma unroll
        for (int t = 0; t < 8; ++t) {
            float x = acc[t][r];
            s += x; sq += x * x;
        }
#pragma unroll
        for (int off = 1; off < 16; off <<= 1) {
            s  += __shfl_xor(s, off, 32);
            sq += __shfl_xor(sq, off, 32);
        }
        float mu = s * (1.f / Dn);
        mean[r] = mu;
        rstd[r] = rsqrtf(sq * (1.f / Dn) - mu * mu + EPS_GN);
    }

    if constexpr (!FUSED) {
#pragma unroll
        for (int t = 0; t < 8; ++t) {
            int col = colbase + t * 16 + lm;
            float g = gamma[col], be = beta[col];
#pragma unroll
            for (int r = 0; r < 8; ++r) {
                int row = mrow + 8 * lh + r;
                vout[(size_t)(r0 + row) * En + col] =
                    (acc[t][r] - mean[r]) * rstd[r] * g + be;
            }
        }
    } else {
        // ---- att = GN(k)*q -> bf16 A-operand tile (permuted-K layout) ----
#pragma unroll
        for (int t = 0; t < 8; ++t) {
            int d = t * 16 + lm;
            int col = colbase + d;
            int dpos = (d & ~31) | permA(d & 31);
            float g = gamma[col], be = beta[col], qv = qg[b * En + col];
#pragma unroll
            for (int r = 0; r < 8; ++r) {
                int row = mrow + 8 * lh + r;
                float y = (acc[t][r] - mean[r]) * rstd[r] * g + be;
                att[row * 136 + dpos] = (bf16_t)(y * qv);
            }
        }
        __syncthreads();  // staging buffers dead -> stage WbT into aliased region
        for (int c = tid; c < 64 * 16; c += 256) {  // 64 cols x 16 chunks of 16B
            int col = c >> 4, ch = c & 15;
            *(uint4*)(Wbs + col * 136 + ch * 8) =
                *(const uint4*)(WbT + col * 128 + ch * 8);
        }
        __syncthreads();

        // ---- GEMM2: basic = relu(att(128x128) @ Wb(128x64) + bb) ----
        v8f acc2[4];
#pragma unroll
        for (int t = 0; t < 4; ++t)
#pragma unroll
            for (int i = 0; i < 8; ++i) acc2[t][i] = 0.f;
#pragma unroll
        for (int kc = 0; kc < 4; ++kc) {
            v16bf a = load_afrag(att + (mrow + lm) * 136 + kc * 32, lh);
#pragma unroll
            for (int tn = 0; tn < 4; ++tn) {
                v16bf bfrag = load_bfrag(Wbs + (tn * 16 + lm) * 136 + kc * 32, lh);
                acc2[tn] = wmma_bf16(a, bfrag, acc2[tn]);
            }
        }
        __syncthreads();  // att dead -> Lb aliases it

#pragma unroll
        for (int tn = 0; tn < 4; ++tn) {
            float bcol = bbv[tn * 16 + lm];
#pragma unroll
            for (int r = 0; r < 8; ++r) {
                float x = acc2[tn][r] + bcol;
                int row = mrow + 8 * lh + r;
                Lb[row * 65 + tn * 16 + lm] = x > 0.f ? x : 0.f;
            }
        }
        __syncthreads();

        // ---- alpha_sp logits: basic @ Wl1 + bl1, masked ----
        if (tid < 128) {
            int m = tid;
            float s = bl1[0];
#pragma unroll 8
            for (int j = 0; j < MIDn; ++j) s += Lb[m * 65 + j] * Wl1[j];
            float mv = mask[b * Mn + m0 + m];
            alpha_out[(size_t)(b * Hn + h) * Mn + m0 + m] = (mv == 0.f) ? -1e9f : s;
        }
        // ---- pool partial: sum_m basic[m,d]*mask[b,m] ----
        if (tid < MIDn) {
            int d = tid;
            float s = 0.f;
            for (int m = 0; m < 128; ++m)
                s += Lb[m * 65 + d] * mask[b * Mn + m0 + m];
            atomicAdd(&pool_sum[(b * Hn + h) * MIDn + d], s);
        }
    }
}

// ---------------------------------------------------------------------------
// softmax over M + v2_agg = sum_m alpha[m] * v2_gn[b,h,m,:]
// ---------------------------------------------------------------------------
__global__ __launch_bounds__(128) void softmax_agg_kernel(
    const float* __restrict__ logits,  // (Bn*Hn, Mn)
    const float* __restrict__ v2g,     // (Bn*Mn, En)
    float* __restrict__ v2agg) {       // (Bn*Hn, Dn)
    const int bh = blockIdx.x, b = bh >> 3, h = bh & 7;
    const int tid = threadIdx.x;
    __shared__ float red[128];
    const float* lg = logits + (size_t)bh * Mn;

    float mx = -3.4e38f;
    for (int m = tid; m < Mn; m += 128) mx = fmaxf(mx, lg[m]);
    red[tid] = mx;
    __syncthreads();
    for (int o = 64; o > 0; o >>= 1) {
        if (tid < o) red[tid] = fmaxf(red[tid], red[tid + o]);
        __syncthreads();
    }
    mx = red[0];
    __syncthreads();

    float s = 0.f;
    for (int m = tid; m < Mn; m += 128) s += __expf(lg[m] - mx);
    red[tid] = s;
    __syncthreads();
    for (int o = 64; o > 0; o >>= 1) {
        if (tid < o) red[tid] += red[tid + o];
        __syncthreads();
    }
    const float inv = 1.f / red[0];

    float acc = 0.f;
    const float* vb = v2g + (size_t)b * Mn * En + h * Dn + tid;
    for (int m = 0; m < Mn; ++m)
        acc += __expf(lg[m] - mx) * inv * vb[(size_t)m * En];
    v2agg[bh * Dn + tid] = acc;
}

// ---------------------------------------------------------------------------
// alpha_ch = sigmoid(pool @ Wl2 + bl2);  out = v1 * v2_agg * alpha_ch
// ---------------------------------------------------------------------------
__global__ __launch_bounds__(128) void finalize_kernel(
    const float* __restrict__ pool_sum, const float* __restrict__ cnt,
    const float* __restrict__ Wl2, const float* __restrict__ bl2,
    const float* __restrict__ v1g, const float* __restrict__ v2agg,
    float* __restrict__ out) {
    const int bh = blockIdx.x, b = bh >> 3, h = bh & 7;
    const int d = threadIdx.x;
    __shared__ float pl[MIDn];
    if (d < MIDn) pl[d] = pool_sum[bh * MIDn + d] / cnt[b];
    __syncthreads();
    float s = bl2[d];
#pragma unroll 8
    for (int j = 0; j < MIDn; ++j) s += pl[j] * Wl2[j * Dn + d];
    float ach = 1.f / (1.f + __expf(-s));
    int e = h * Dn + d;
    out[b * En + e] = v1g[b * En + e] * v2agg[bh * Dn + d] * ach;
}

// ---------------------------------------------------------------------------
extern "C" void kernel_launch(void* const* d_in, const int* in_sizes, int n_in,
                              void* d_out, int out_size, void* d_ws, size_t ws_size,
                              hipStream_t stream) {
    const float* query  = (const float*)d_in[0];
    const float* key    = (const float*)d_in[1];
    const float* mask   = (const float*)d_in[2];
    const float* value1 = (const float*)d_in[3];
    const float* value2 = (const float*)d_in[4];
    const float* Wq  = (const float*)d_in[5];
    const float* bq  = (const float*)d_in[6];
    const float* gq  = (const float*)d_in[7];
    const float* Bq  = (const float*)d_in[8];
    const float* Wk  = (const float*)d_in[9];
    const float* bk  = (const float*)d_in[10];
    const float* gk  = (const float*)d_in[11];
    const float* Bk  = (const float*)d_in[12];
    const float* Wv1 = (const float*)d_in[13];
    const float* bv1 = (const float*)d_in[14];
    const float* gv1 = (const float*)d_in[15];
    const float* Bv1 = (const float*)d_in[16];
    const float* Wv2 = (const float*)d_in[17];
    const float* bv2 = (const float*)d_in[18];
    const float* gv2 = (const float*)d_in[19];
    const float* Bv2 = (const float*)d_in[20];
    const float* Wb  = (const float*)d_in[21];
    const float* bb  = (const float*)d_in[22];
    const float* Wl1 = (const float*)d_in[23];
    const float* bl1 = (const float*)d_in[24];
    const float* Wl2 = (const float*)d_in[25];
    const float* bl2 = (const float*)d_in[26];

    char* ws = (char*)d_ws;
    size_t off = 0;
    auto take = [&](size_t bytes) {
        char* p = ws + off;
        off += (bytes + 255) & ~(size_t)255;
        return p;
    };
    bf16_t* WqT  = (bf16_t*)take((size_t)En * En * 2);
    bf16_t* WkT  = (bf16_t*)take((size_t)En * En * 2);
    bf16_t* Wv1T = (bf16_t*)take((size_t)En * En * 2);
    bf16_t* Wv2T = (bf16_t*)take((size_t)En * En * 2);
    bf16_t* WbT  = (bf16_t*)take((size_t)Dn * MIDn * 2);
    float* QG    = (float*)take((size_t)Bn * En * 4);
    float* V1G   = (float*)take((size_t)Bn * En * 4);
    float* V2G   = (float*)take((size_t)Bn * Mn * En * 4);   // 134 MB
    float* ALPHA = (float*)take((size_t)Bn * Hn * Mn * 4);
    float* POOL  = (float*)take((size_t)Bn * Hn * MIDn * 4);
    float* CNT   = (float*)take((size_t)Bn * 4);
    float* V2AGG = (float*)take((size_t)Bn * Hn * Dn * 4);

    const int nW = En * En;
    cvt_T_bf16_kernel<<<(nW + 255) / 256, 256, 0, stream>>>(Wq, WqT, En, En);
    cvt_T_bf16_kernel<<<(nW + 255) / 256, 256, 0, stream>>>(Wk, WkT, En, En);
    cvt_T_bf16_kernel<<<(nW + 255) / 256, 256, 0, stream>>>(Wv1, Wv1T, En, En);
    cvt_T_bf16_kernel<<<(nW + 255) / 256, 256, 0, stream>>>(Wv2, Wv2T, En, En);
    cvt_T_bf16_kernel<<<(Dn * MIDn + 255) / 256, 256, 0, stream>>>(Wb, WbT, Dn, MIDn);

    init_kernel<<<Bn, 256, 0, stream>>>(mask, POOL, CNT);

    proj_small_kernel<<<16, 256, 0, stream>>>(query, value1, WqT, Wv1T,
                                              bq, bv1, gq, gv1, Bq, Bv1, QG, V1G);

    const size_t shbig = 34816 + 4 * 10240;  // att/Lb + double-buffered As/Bs
    proj_big_kernel<true><<<dim3(Bn * Mn / 128, Hn), 256, shbig, stream>>>(
        key, WkT, bk, gk, Bk, QG, mask, WbT, bb, Wl1, bl1, ALPHA, POOL, nullptr);
    proj_big_kernel<false><<<dim3(Bn * Mn / 128, Hn), 256, shbig, stream>>>(
        value2, Wv2T, bv2, gv2, Bv2, nullptr, nullptr, nullptr, nullptr,
        nullptr, nullptr, nullptr, nullptr, V2G);

    softmax_agg_kernel<<<Bn * Hn, 128, 0, stream>>>(ALPHA, V2G, V2AGG);
    finalize_kernel<<<Bn * Hn, 128, 0, stream>>>(POOL, CNT, Wl2, bl2, V1G, V2AGG,
                                                 (float*)d_out);
}
